// GNNNewsRecommender_678604832877
// MI455X (gfx1250) — matrix operand order repaired and checked by reading, not compiled, not run on hardware
//
#include <hip/hip_runtime.h>
#include <hip/hip_bf16.h>
#include <math.h>

// ---------------------------------------------------------------------------
// GNN news recommender, fused for MI455X (gfx1250, wave32, WMMA).
//   B=8192, H=50, D=128. GEMMs via v_wmma_f32_16x16x32_bf16.
// ---------------------------------------------------------------------------

typedef __attribute__((ext_vector_type(16))) __bf16        v16bf;
typedef __attribute__((ext_vector_type(8)))  float         v8f;
typedef __attribute__((ext_vector_type(4)))  unsigned int  u32x4;

union FragU { v16bf v; u32x4 q[2]; };

#define XS  272   // X row stride (bf16), 64 rows x 256 cols + pad (breaks 512B bank pattern)
#define ES  132   // E row stride (f32), 128 cols + pad
#define EBS 136   // Ebf row stride (bf16), 128 cols + pad
#define R0  34816 // bytes: max(64*XS*2, 64*ES*4)
#define R1  17408 // bytes: 64*EBS*2

__device__ __forceinline__ unsigned short f2bf(float f) {
    unsigned int u = __builtin_bit_cast(unsigned int, f);
    u += 0x7FFFu + ((u >> 16) & 1u);          // round-to-nearest-even
    return (unsigned short)(u >> 16);
}

// A fragment (16x32 bf16, row-major in LDS). Per ISA layout:
// lanes 0-15: K 0-7 (vgpr0-3) + K 16-23 (vgpr4-7); lanes 16-31: +8.
__device__ __forceinline__ v16bf ldfragA(const unsigned short* base, int stride,
                                         int mt, int kt, int lane) {
    int m  = (mt << 4) + (lane & 15);
    int ah = (lane >> 4) << 3;
    const unsigned short* p = base + m * stride + (kt << 5) + ah;
    FragU f;
    f.q[0] = *(const u32x4*)p;          // K = base .. base+7
    f.q[1] = *(const u32x4*)(p + 16);   // K = base+16 .. base+23
    return f.v;
}

// B fragment (32x16 bf16). Weights stored N-major (transposed), so each lane
// reads two 16-byte chunks: lane = column N, K 0-7 + K 16-23; lanes 16-31: +8.
__device__ __forceinline__ v16bf ldfragB(const unsigned short* wT, int stride,
                                         int n0, int kt, int lane) {
    int n  = n0 + (lane & 15);
    int bh = (lane >> 4) << 3;
    const unsigned short* p = wT + n * stride + (kt << 5) + bh;
    FragU f;
    f.q[0] = *(const u32x4*)p;
    f.q[1] = *(const u32x4*)(p + 16);
    return f.v;
}

#define WMMA_BF16(a, b, c) \
    __builtin_amdgcn_wmma_f32_16x16x32_bf16(false, (a), false, (b), (short)0, (c), false, false)

// Gather 64 rows of concat(news_table[idx], gnn[idx] masked) into LDS X (bf16).
__device__ __forceinline__ void gather_rows(unsigned short* Xs, const int* idxs,
                                            int nrows, const float* news_table,
                                            const float* gnn, int num_gnn, int tid) {
    int sub  = tid >> 6;          // 0..3 : row within 4-row iteration
    int part = (tid >> 5) & 1;    // 0 = id embedding, 1 = gnn embedding
    int lane = tid & 31;          // float4 slot
    for (int lr = sub; lr < 64; lr += 4) {
        float4 val = make_float4(0.f, 0.f, 0.f, 0.f);
        if (lr < nrows) {
            int idx = idxs[lr];
            if (part == 0) {
                val = *(const float4*)(news_table + (long long)idx * 128 + lane * 4);
            } else if (idx < num_gnn) {
                val = *(const float4*)(gnn + (long long)idx * 128 + lane * 4);
            }
        }
        unsigned short* dst = Xs + lr * XS + part * 128 + lane * 4;
        unsigned int lo = (unsigned int)f2bf(val.x) | ((unsigned int)f2bf(val.y) << 16);
        unsigned int hi = (unsigned int)f2bf(val.z) | ((unsigned int)f2bf(val.w) << 16);
        *(unsigned int*)dst       = lo;
        *(unsigned int*)(dst + 2) = hi;
    }
}

// Fusion: E = relu(LN(X @ fusion_w + b)); writes E (f32) over the X region and
// a bf16 copy Ebf for the following GEMM. 8 waves, 1 N-tile each, 4 M-tiles
// held in registers so X stays valid for the whole GEMM.
__device__ __forceinline__ void fusion_block(unsigned short* Xs, float* Es,
                                             unsigned short* Ebs,
                                             const unsigned short* fwT, const float* fb,
                                             const float* lng, const float* lnb, int tid) {
    const int lane = tid & 31, wave = tid >> 5;
    const int n0 = wave << 4;
    v8f acc[4];
    #pragma unroll
    for (int i = 0; i < 4; ++i) acc[i] = (v8f){0, 0, 0, 0, 0, 0, 0, 0};
    #pragma unroll
    for (int kt = 0; kt < 8; ++kt) {                 // K = 256
        v16bf b = ldfragB(fwT, 256, n0, kt, lane);
        #pragma unroll
        for (int mt = 0; mt < 4; ++mt) {
            v16bf a = ldfragA(Xs, XS, mt, kt, lane);
            acc[mt] = WMMA_BF16(a, b, acc[mt]);
        }
    }
    __syncthreads();                                 // all reads of X done; reuse as E
    const int nn  = n0 + (lane & 15);
    const int hi8 = (lane >> 4) << 3;
    float bias = fb[nn];
    #pragma unroll
    for (int mt = 0; mt < 4; ++mt)
        #pragma unroll
        for (int v = 0; v < 8; ++v) {
            int m = (mt << 4) + hi8 + v;
            Es[m * ES + nn] = acc[mt][v] + bias;
        }
    __syncthreads();
    // LayerNorm + ReLU: 4 threads per row (contiguous lanes -> shfl_xor works)
    {
        int r = tid >> 2, q = tid & 3;
        float* row = Es + r * ES + (q << 5);
        float s = 0.f, ss = 0.f;
        #pragma unroll
        for (int i = 0; i < 32; ++i) { float x = row[i]; s += x; ss += x * x; }
        s  += __shfl_xor(s, 1, 32);  ss += __shfl_xor(ss, 1, 32);
        s  += __shfl_xor(s, 2, 32);  ss += __shfl_xor(ss, 2, 32);
        float mean = s * (1.f / 128.f);
        float var  = ss * (1.f / 128.f) - mean * mean;
        float rinv = rsqrtf(var + 1e-5f);
        unsigned short* brow = Ebs + r * EBS + (q << 5);
        #pragma unroll
        for (int i = 0; i < 32; ++i) {
            int n = (q << 5) + i;
            float y = (row[i] - mean) * rinv * lng[n] + lnb[n];
            y = fmaxf(y, 0.f);
            row[i]  = y;
            brow[i] = f2bf(y);
        }
    }
    __syncthreads();
}

// ---------------------------------------------------------------------------
// Kernel 0: fp32 weights -> bf16, transposed to N-major for B-fragment loads.
__global__ __launch_bounds__(256) void convert_weights(const float* fw, const float* a1w,
                                                       const float* ntw,
                                                       unsigned short* fT,
                                                       unsigned short* aT,
                                                       unsigned short* nT) {
    int i = blockIdx.x * 256 + threadIdx.x;
    if (i < 32768)       { int n = i >> 8, k = i & 255;                 fT[i] = f2bf(fw[k * 128 + n]); }
    else if (i < 40960)  { int j = i - 32768; int n = j >> 7, k = j & 127; aT[j] = f2bf(a1w[k * 64 + n]); }
    else if (i < 57344)  { int j = i - 40960; int n = j >> 7, k = j & 127; nT[j] = f2bf(ntw[k * 128 + n]); }
}

// ---------------------------------------------------------------------------
// Kernel 1: news_repr = relu(fusion(concat(id,gnn)) @ nt_w + nt_b), 64 rows/block.
__global__ __launch_bounds__(256) void news_repr_kernel(
        const int* news_idx, const float* news_table, const float* gnn,
        const unsigned short* fwT, const float* fb, const float* lng, const float* lnb,
        const unsigned short* ntT, const float* ntb,
        float* NR, int Bsz, int num_gnn) {
    __shared__ __align__(16) unsigned char smem[R0 + R1 + 256];
    unsigned short* Xs  = (unsigned short*)smem;
    float*          Es  = (float*)smem;
    unsigned short* Ebs = (unsigned short*)(smem + R0);
    int*            idxs = (int*)(smem + R0 + R1);

    int tid = threadIdx.x;
    int base = blockIdx.x * 64;
    int nrows = min(64, Bsz - base);
    if (tid < 64) idxs[tid] = (tid < nrows) ? news_idx[base + tid] : 0;
    __syncthreads();
    gather_rows(Xs, idxs, nrows, news_table, gnn, num_gnn, tid);
    __syncthreads();
    fusion_block(Xs, Es, Ebs, fwT, fb, lng, lnb, tid);

    // second GEMM: relu(Ebf(64x128) @ nt_wT + nt_b)
    int lane = tid & 31, wave = tid >> 5;
    int n0 = wave << 4;
    v8f acc[4];
    #pragma unroll
    for (int i = 0; i < 4; ++i) acc[i] = (v8f){0, 0, 0, 0, 0, 0, 0, 0};
    #pragma unroll
    for (int kt = 0; kt < 4; ++kt) {                 // K = 128
        v16bf b = ldfragB(ntT, 128, n0, kt, lane);
        #pragma unroll
        for (int mt = 0; mt < 4; ++mt) {
            v16bf a = ldfragA(Ebs, EBS, mt, kt, lane);
            acc[mt] = WMMA_BF16(a, b, acc[mt]);
        }
    }
    int nn = n0 + (lane & 15), hi8 = (lane >> 4) << 3;
    float bias = ntb[nn];
    #pragma unroll
    for (int mt = 0; mt < 4; ++mt)
        #pragma unroll
        for (int v = 0; v < 8; ++v) {
            int m = (mt << 4) + hi8 + v;
            if (m < nrows)
                NR[(long long)(base + m) * 128 + nn] = fmaxf(acc[mt][v] + bias, 0.f);
        }
}

// ---------------------------------------------------------------------------
// Kernel 2: one batch row per block: history fusion -> attention -> user repr
//           -> dot(news_repr) -> sigmoid.
__global__ __launch_bounds__(256) void user_kernel(
        const int* user_idx, const int* history,
        const float* user_table, const float* news_table, const float* gnn,
        const unsigned short* fwT, const float* fb, const float* lng, const float* lnb,
        const unsigned short* a1T, const float* a1b, const float* a2w, const float* a2b,
        const float* utw, const float* utb,
        const float* NR, float* out, int num_gnn) {
    __shared__ __align__(16) unsigned char smem[R0 + R1 + 1536];
    unsigned short* Xs     = (unsigned short*)smem;
    float*          Es     = (float*)smem;
    unsigned short* Ebs    = (unsigned short*)(smem + R0);
    int*            hist_s = (int*)  (smem + R0 + R1);
    float*          score  = (float*)(smem + R0 + R1 + 256);
    float*          attn   = (float*)(smem + R0 + R1 + 512);
    float*          u_s    = (float*)(smem + R0 + R1 + 768);
    float*          red    = (float*)(smem + R0 + R1 + 1280);

    int tid = threadIdx.x;
    int b   = blockIdx.x;
    const int* hrow = history + (long long)b * 50;
    if (tid < 64) hist_s[tid] = (tid < 50) ? hrow[tid] : 0;
    if (tid >= 64 && tid < 128) score[tid - 64] = a2b[0];
    __syncthreads();
    gather_rows(Xs, hist_s, 50, news_table, gnn, num_gnn, tid);
    __syncthreads();
    fusion_block(Xs, Es, Ebs, fwT, fb, lng, lnb, tid);

    // attention GEMM: tanh(Ebf(64x128) @ attn1_wT(64x128 N-major) + b1) . attn2_w
    int lane = tid & 31, wave = tid >> 5;
    #pragma unroll
    for (int half = 0; half < 2; ++half) {
        int t2 = wave + half * 8;                    // 16 tiles over 8 waves
        int mt = t2 >> 2, nt = t2 & 3;
        v8f acc = (v8f){0, 0, 0, 0, 0, 0, 0, 0};
        #pragma unroll
        for (int kt = 0; kt < 4; ++kt) {             // K = 128
            v16bf a  = ldfragA(Ebs, EBS, mt, kt, lane);
            v16bf bb = ldfragB(a1T, 128, nt << 4, kt, lane);
            acc = WMMA_BF16(a, bb, acc);
        }
        int nn  = (nt << 4) + (lane & 15);
        float b1 = a1b[nn], w2 = a2w[nn];
        int hi8 = (lane >> 4) << 3;
        #pragma unroll
        for (int v = 0; v < 8; ++v) {
            int m = (mt << 4) + hi8 + v;
            float val = tanhf(acc[v] + b1);
            atomicAdd(&score[m], val * w2);          // ds_add_f32
        }
    }
    __syncthreads();

    // masked softmax over H=50 (history==0 -> -1e9), serial by thread 0 (tiny)
    if (tid == 0) {
        float mx = -1e30f;
        for (int h = 0; h < 50; ++h) {
            float s = (hist_s[h] != 0) ? score[h] : -1e9f;
            score[h] = s;
            mx = fmaxf(mx, s);
        }
        float sum = 0.f;
        for (int h = 0; h < 50; ++h) { float e = expf(score[h] - mx); attn[h] = e; sum += e; }
        float inv = 1.f / sum;
        for (int h = 0; h < 50; ++h) attn[h] *= inv;
    }
    __syncthreads();

    // weighted sum + user embedding
    if (tid < 128) {
        float hr = 0.f;
        for (int h = 0; h < 50; ++h) hr += attn[h] * Es[h * ES + tid];
        u_s[tid] = user_table[(long long)user_idx[b] * 128 + tid] + hr;
    }
    __syncthreads();

    // user transform (1x128 @ 128x128, VALU — WMMA would waste 15/16 of the tile)
    float p = 0.f;
    if (tid < 128) {
        float acc = utb[tid];
        for (int k = 0; k < 128; ++k) acc = fmaf(u_s[k], utw[k * 128 + tid], acc);
        float ur = fmaxf(acc, 0.f);
        p = ur * NR[(long long)b * 128 + tid];
    }
    p += __shfl_xor(p, 1, 32);  p += __shfl_xor(p, 2, 32);  p += __shfl_xor(p, 4, 32);
    p += __shfl_xor(p, 8, 32);  p += __shfl_xor(p, 16, 32);
    if (lane == 0) red[wave] = p;
    __syncthreads();
    if (tid == 0) {
        float d = red[0] + red[1] + red[2] + red[3] + red[4] + red[5] + red[6] + red[7];
        out[b] = 1.f / (1.f + expf(-d));
    }
}

// ---------------------------------------------------------------------------
extern "C" void kernel_launch(void* const* d_in, const int* in_sizes, int n_in,
                              void* d_out, int out_size, void* d_ws, size_t ws_size,
                              hipStream_t stream) {
    const int*   user_idx   = (const int*)d_in[0];
    const int*   news_idx   = (const int*)d_in[1];
    const int*   history    = (const int*)d_in[2];
    const float* gnn        = (const float*)d_in[3];
    const float* user_table = (const float*)d_in[4];
    const float* news_table = (const float*)d_in[5];
    const float* fusion_w   = (const float*)d_in[6];
    const float* fusion_b   = (const float*)d_in[7];
    const float* ln_g       = (const float*)d_in[8];
    const float* ln_b       = (const float*)d_in[9];
    const float* attn1_w    = (const float*)d_in[10];
    const float* attn1_b    = (const float*)d_in[11];
    const float* attn2_w    = (const float*)d_in[12];
    const float* attn2_b    = (const float*)d_in[13];
    const float* ut_w       = (const float*)d_in[14];
    const float* ut_b       = (const float*)d_in[15];
    const float* nt_w       = (const float*)d_in[16];
    const float* nt_b       = (const float*)d_in[17];

    int B = in_sizes[0];
    int num_gnn = in_sizes[3] / 128;

    // workspace layout: [0,112KB) bf16 weights, [128KB, +4MB) news_repr f32
    unsigned short* fwT = (unsigned short*)d_ws;
    unsigned short* a1T = fwT + 32768;
    unsigned short* ntT = a1T + 8192;
    float* NR  = (float*)((char*)d_ws + (128 << 10));
    float* out = (float*)d_out;

    convert_weights<<<224, 256, 0, stream>>>(fusion_w, attn1_w, nt_w, fwT, a1T, ntT);
    news_repr_kernel<<<(B + 63) / 64, 256, 0, stream>>>(
        news_idx, news_table, gnn, fwT, fusion_b, ln_g, ln_b, ntT, nt_b, NR, B, num_gnn);
    user_kernel<<<B, 256, 0, stream>>>(
        user_idx, history, user_table, news_table, gnn,
        fwT, fusion_b, ln_g, ln_b, a1T, attn1_b, attn2_w, attn2_b,
        ut_w, ut_b, NR, out, num_gnn);
}
